// GroupedQueryAttention_59889023976046
// MI455X (gfx1250) — matrix operand order repaired and checked by reading, not compiled, not run on hardware
//
#include <hip/hip_runtime.h>
#include <stdint.h>
#include <stddef.h>

// ---------------- problem constants ----------------
#define BATCH   2
#define NSEQ    2048
#define DIMM    2048
#define KDIMM   2048
#define QHEADS  16
#define KVHEADS 4
#define HEADDIM 128
#define KVDIM   512
#define MROWS   (BATCH * NSEQ)   // 4096

// LDS tile geometry: 128 rows x 32 bf16, padded to 40 bf16/row (80 B)
// -> bank stride 20 DWORDs, conflict-free for 16-lane fragment reads.
#define TROW    40
#define TSIZE   (128 * TROW)

// ---------------- vector types ----------------
typedef __attribute__((ext_vector_type(16))) __bf16  v16bf;
typedef __attribute__((ext_vector_type(8)))  __bf16  v8bf;
typedef __attribute__((ext_vector_type(8)))  float   v8f;
typedef __attribute__((ext_vector_type(4)))  unsigned v4u;
typedef __attribute__((ext_vector_type(8)))  unsigned v8u;
typedef __attribute__((ext_vector_type(4)))  int      v4i;
typedef __attribute__((ext_vector_type(8)))  int      v8i;

// ---------------- TDM availability ----------------
#if defined(__has_builtin)
#if __has_builtin(__builtin_amdgcn_tensor_load_to_lds)
#define USE_TDM 1
#endif
#endif
#ifndef USE_TDM
#define USE_TDM 0
#endif

#if defined(__has_include)
#if __has_include(<hip/amd_detail/amd_gfx1250_TDM.h>)
#define TDM_ARGS6 1
#endif
#endif
#ifndef TDM_ARGS6
#define TDM_ARGS6 0
#endif

// ---------------- helpers ----------------
__device__ __forceinline__ v16bf cat8(v8bf lo, v8bf hi) {
    return __builtin_shufflevector(lo, hi, 0, 1, 2, 3, 4, 5, 6, 7,
                                           8, 9, 10, 11, 12, 13, 14, 15);
}

__device__ __forceinline__ v8f wmma_bf16(v16bf a, v16bf b, v8f c) {
    // D = A(16x32 bf16) x B(32x16 bf16) + C(16x16 f32)
    return __builtin_amdgcn_wmma_f32_16x16x32_bf16(
        /*neg_a=*/false, a, /*neg_b=*/false, b,
        /*c_mod=*/(short)0, c, /*reuse_a=*/false, /*reuse_b=*/false);
}

#if USE_TDM
// 2-D tile load via the Tensor Data Mover (D# per CDNA5 ISA ch.8).
// Loads tile1 rows x 32 bf16 into LDS; D# pad fields insert 8 bf16 of
// padding after each 64 B row -> LDS row stride TROW (=40) elements.
__device__ __forceinline__ void tdm_load_2d(unsigned ldsOff, const void* gptr,
                                            unsigned dim0, unsigned dim1,
                                            unsigned tile0, unsigned tile1,
                                            unsigned stride0) {
    unsigned long long ga = (unsigned long long)(size_t)gptr;
    v4u g0;
    g0.x = 1u;                                    // count=1, user descriptor
    g0.y = ldsOff;                                // LDS byte address
    g0.z = (unsigned)(ga & 0xffffffffull);        // global_addr[31:0]
    g0.w = (unsigned)((ga >> 32) & 0x1ffffffull)  // global_addr[56:32]
           | (2u << 30);                          // type = 2 ("image")
    v8u g1u;
    g1u[0] = 0x00010000u                          // wg_mask=0, data_size=1 (2B)
             | (1u << 20)                         // pad_enable
             | (3u << 22)                         // pad_interval: 16 DWORDs
             | (3u << 25);                        // pad_amount: 4 DWORDs
    g1u[1] = (dim0 & 0xffffu) << 16;              // tensor_dim0 lo16
    g1u[2] = (dim0 >> 16) | ((dim1 & 0xffffu) << 16);
    g1u[3] = (dim1 >> 16) | ((tile0 & 0xffffu) << 16);
    g1u[4] = (tile1 & 0xffffu);                   // tile_dim1, tile_dim2=0
    g1u[5] = stride0;                             // tensor_dim0_stride lo32
    g1u[6] = 0u;                                  // stride0 hi16, stride1 lo16
    g1u[7] = 0u;                                  // stride1 hi32
    v8i g1 = __builtin_bit_cast(v8i, g1u);
    v4i gz = {0, 0, 0, 0};
#if TDM_ARGS6
    v8i gz8 = {0, 0, 0, 0, 0, 0, 0, 0};
    __builtin_amdgcn_tensor_load_to_lds(g0, g1, gz, gz, gz8, 0);
#else
    __builtin_amdgcn_tensor_load_to_lds(g0, g1, gz, gz, 0);
#endif
}
#endif

// ---------------- fp32 -> bf16 conversion ----------------
__global__ void cvt_bf16(const float* __restrict__ src,
                         __bf16* __restrict__ dst, int n) {
    int i = blockIdx.x * blockDim.x + threadIdx.x;
    int stride = gridDim.x * blockDim.x;
    for (; i < n; i += stride) dst[i] = (__bf16)src[i];
}

// ---------------- fused GEMM:  Y = X @ W^T + b (+ epilogue) ----------------
// X: [M x Kdim] bf16 row-major, W: [Nout x Kdim] bf16 row-major.
// MODE_Q / MODE_K : RoPE epilogue, bf16 row-major out
// MODE_V          : transposed bf16 out -> VT[b][d][s]
// MODE_O          : f32 row-major out (final result)
enum { MODE_Q = 0, MODE_K = 1, MODE_V = 2, MODE_O = 3 };

template <int MODE>
__global__ __launch_bounds__(128) void gemm_kernel(
    const __bf16* __restrict__ X, const __bf16* __restrict__ W,
    const float* __restrict__ bias, void* __restrict__ outp,
    int Nout, int Kdim) {
    __shared__ __align__(32) __bf16 Xs[2][TSIZE];   // double-buffered tiles
    __shared__ __align__(32) __bf16 Ws[2][TSIZE];

    const int tid  = threadIdx.x;
    const int wv   = tid >> 5;
    const int lane = tid & 31;
    const int half = lane >> 4;
    const int ln   = lane & 15;
    const int wm   = wv >> 1;        // 2x2 wave grid, each wave 64x64
    const int wn   = wv & 1;
    const int m0   = blockIdx.y * 128;
    const int n0   = blockIdx.x * 128;
    const int nk   = Kdim >> 5;      // 32-wide k-steps

    v8f acc[4][4] = {};

#if USE_TDM
    // prologue: issue k-step 0 into buffer 0
    if (tid == 0) {
        tdm_load_2d((unsigned)(size_t)&Xs[0][0], X + (size_t)m0 * Kdim,
                    (unsigned)Kdim, (unsigned)MROWS, 32u, 128u, (unsigned)Kdim);
        tdm_load_2d((unsigned)(size_t)&Ws[0][0], W + (size_t)n0 * Kdim,
                    (unsigned)Kdim, (unsigned)Nout, 32u, 128u, (unsigned)Kdim);
    }
#endif

    for (int ki = 0; ki < nk; ++ki) {
        const int cur = ki & 1;
#if USE_TDM
        if (tid == 0) {
            if (ki + 1 < nk) {
                // software pipeline: prefetch next tile, then wait for current.
                // Tensor ops complete in order -> tensorcnt<=2 means the pair
                // for step ki has landed while step ki+1 is still in flight.
                const int k1 = (ki + 1) * 32;
                tdm_load_2d((unsigned)(size_t)&Xs[cur ^ 1][0],
                            X + (size_t)m0 * Kdim + k1, (unsigned)Kdim,
                            (unsigned)MROWS, 32u, 128u, (unsigned)Kdim);
                tdm_load_2d((unsigned)(size_t)&Ws[cur ^ 1][0],
                            W + (size_t)n0 * Kdim + k1, (unsigned)Kdim,
                            (unsigned)Nout, 32u, 128u, (unsigned)Kdim);
                __builtin_amdgcn_s_wait_tensorcnt((short)2);
            } else {
                __builtin_amdgcn_s_wait_tensorcnt((short)0);
            }
        }
#else
        {
            const int k0 = ki * 32;
            const __bf16* sx = X + (size_t)(m0 + tid) * Kdim + k0;
            const __bf16* sw = W + (size_t)(n0 + tid) * Kdim + k0;
            uint4* dx = (uint4*)&Xs[cur][tid * TROW];
            uint4* dw = (uint4*)&Ws[cur][tid * TROW];
#pragma unroll
            for (int c = 0; c < 4; ++c) {
                dx[c] = ((const uint4*)sx)[c];
                dw[c] = ((const uint4*)sw)[c];
            }
        }
#endif
        __syncthreads();

        const __bf16* xb = &Xs[cur][0];
        const __bf16* wb = &Ws[cur][0];
        v16bf af[4], bfr[4];
#pragma unroll
        for (int mt = 0; mt < 4; ++mt) {
            int r = (wm * 64 + mt * 16 + ln) * TROW;
            af[mt] = cat8(*(const v8bf*)&xb[r + half * 16],
                          *(const v8bf*)&xb[r + half * 16 + 8]);
        }
#pragma unroll
        for (int nt = 0; nt < 4; ++nt) {
            int r = (wn * 64 + nt * 16 + ln) * TROW + half * 16;
            bfr[nt] = cat8(*(const v8bf*)&wb[r], *(const v8bf*)&wb[r + 8]);
        }
#pragma unroll
        for (int mt = 0; mt < 4; ++mt)
#pragma unroll
            for (int nt = 0; nt < 4; ++nt)
                acc[mt][nt] = wmma_bf16(af[mt], bfr[nt], acc[mt][nt]);

        __syncthreads();
    }

    // ---------------- epilogue ----------------
#pragma unroll
    for (int mt = 0; mt < 4; ++mt) {
#pragma unroll
        for (int nt = 0; nt < 4; ++nt) {
            const int col  = n0 + wn * 64 + nt * 16 + ln;
            const int rowb = m0 + wm * 64 + mt * 16;
            const float bia = bias[col];
            v8f a = acc[mt][nt];

            if constexpr (MODE == MODE_Q || MODE == MODE_K) {
                // fused RoPE: pairs are (even col, odd col) = (lane, lane^1)
                const int pairi = (col & (HEADDIM - 1)) >> 1;
                const float invf =
                    __powf(10000.0f, -(float)pairi * (1.0f / 64.0f));
                const bool isreal = (col & 1) == 0;
                __bf16* dst = (__bf16*)outp;
#pragma unroll
                for (int j = 0; j < 8; ++j) {
                    int row = rowb + j + 8 * half;
                    float v = a[j] + bia;
                    float partner = __shfl_xor(v, 1);
                    float sn, cs;
                    __sincosf((float)(row & (NSEQ - 1)) * invf, &sn, &cs);
                    float o = isreal ? (v * cs - partner * sn)
                                     : (partner * sn + v * cs);
                    dst[(size_t)row * Nout + col] = (__bf16)o;
                }
            } else if constexpr (MODE == MODE_V) {
                // store transposed: VT[b][dcol][s], 8 consecutive s per lane
                __bf16* vt = (__bf16*)outp;
                int srow = rowb + 8 * half;
                int bidx = srow >> 11;
                int s    = srow & (NSEQ - 1);
                v8bf pk;
#pragma unroll
                for (int j = 0; j < 8; ++j) pk[j] = (__bf16)(a[j] + bia);
                *(v8bf*)&vt[((size_t)bidx * KVDIM + col) * NSEQ + s] = pk;
            } else {  // MODE_O : f32 result
                float* dst = (float*)outp;
#pragma unroll
                for (int j = 0; j < 8; ++j) {
                    int row = rowb + j + 8 * half;
                    dst[(size_t)row * Nout + col] = a[j] + bia;
                }
            }
        }
    }
}

// ---------------- causal GQA flash attention ----------------
// grid: (NSEQ/64, QHEADS, BATCH); block: 128 (4 waves, 16 q-rows each)
__global__ __launch_bounds__(128) void attn_kernel(
    const __bf16* __restrict__ Q, const __bf16* __restrict__ K,
    const __bf16* __restrict__ VT, __bf16* __restrict__ CTX) {
    // per-wave P staging (D-layout -> A-layout), rows padded to 40 elements
    __shared__ __align__(32) __bf16 pbuf[4][16][TROW];

    const int tid  = threadIdx.x;
    const int wv   = tid >> 5;
    const int lane = tid & 31;
    const int half = lane >> 4;
    const int ln   = lane & 15;
    const int qblk = blockIdx.x;
    const int h    = blockIdx.y;
    const int b    = blockIdx.z;
    const int kvh  = h >> 2;                  // 4 q-heads per kv-head
    const int q0   = qblk * 64 + wv * 16;

    // Q fragments (16 rows x 128 head-dim), resident whole kernel
    v16bf qf[4];
    {
        const __bf16* qr =
            Q + ((size_t)(b * NSEQ + q0 + ln)) * DIMM + h * HEADDIM;
#pragma unroll
        for (int kc = 0; kc < 4; ++kc)
            qf[kc] = cat8(*(const v8bf*)(qr + kc * 32 + half * 16),
                          *(const v8bf*)(qr + kc * 32 + half * 16 + 8));
    }

    v8f o[8] = {};
    float mst[8], lst[8];
#pragma unroll
    for (int j = 0; j < 8; ++j) { mst[j] = -3.0e38f; lst[j] = 0.0f; }
    const float scale = 0.08838834764831845f;  // 1/sqrt(128)
    const int qlast = q0 + 15;

    for (int s0 = 0; s0 <= qlast; s0 += 32) {
        // ---- scores: two 16x16 tiles over 32 key positions ----
        v8f S0 = {}, S1 = {};
#pragma unroll
        for (int kc = 0; kc < 4; ++kc) {
            const __bf16* kr0 = K + ((size_t)(b * NSEQ + s0 + ln)) * KVDIM +
                                kvh * HEADDIM + kc * 32 + half * 16;
            const __bf16* kr1 = kr0 + (size_t)16 * KVDIM;
            S0 = wmma_bf16(qf[kc], *(const v16bf*)kr0, S0);
            S1 = wmma_bf16(qf[kc], *(const v16bf*)kr1, S1);
        }

        // ---- causal mask + online softmax ----
        float bm[8], sc[8], rs[8], p0[8], p1[8];
#pragma unroll
        for (int j = 0; j < 8; ++j) {
            int qg = q0 + j + 8 * half;
            float x0 = (s0 + ln <= qg) ? S0[j] * scale : -3.0e38f;
            float x1 = (s0 + 16 + ln <= qg) ? S1[j] * scale : -3.0e38f;
            S0[j] = x0; S1[j] = x1;
            bm[j] = fmaxf(x0, x1);
        }
#pragma unroll
        for (int msk = 8; msk >= 1; msk >>= 1)
#pragma unroll
            for (int j = 0; j < 8; ++j)
                bm[j] = fmaxf(bm[j], __shfl_xor(bm[j], msk));
#pragma unroll
        for (int j = 0; j < 8; ++j) {
            float mnew = fmaxf(mst[j], bm[j]);
            sc[j] = __expf(mst[j] - mnew);
            p0[j] = (S0[j] > -1.0e37f) ? __expf(S0[j] - mnew) : 0.0f;
            p1[j] = (S1[j] > -1.0e37f) ? __expf(S1[j] - mnew) : 0.0f;
            rs[j] = p0[j] + p1[j];
            mst[j] = mnew;
        }
#pragma unroll
        for (int msk = 8; msk >= 1; msk >>= 1)
#pragma unroll
            for (int j = 0; j < 8; ++j) rs[j] += __shfl_xor(rs[j], msk);
#pragma unroll
        for (int j = 0; j < 8; ++j) lst[j] = lst[j] * sc[j] + rs[j];
#pragma unroll
        for (int t = 0; t < 8; ++t)
#pragma unroll
            for (int j = 0; j < 8; ++j) o[t][j] *= sc[j];

        // ---- D-layout -> A-layout transpose of P through per-wave LDS ----
#pragma unroll
        for (int j = 0; j < 8; ++j) {
            pbuf[wv][j + 8 * half][ln]      = (__bf16)p0[j];
            pbuf[wv][j + 8 * half][16 + ln] = (__bf16)p1[j];
        }
        asm volatile("s_wait_dscnt 0x0" ::: "memory");  // same-wave ds ordering
        v16bf pf = cat8(*(const v8bf*)&pbuf[wv][ln][half * 16],
                        *(const v8bf*)&pbuf[wv][ln][half * 16 + 8]);

        // ---- O += P @ V  (VT rows contiguous along s) ----
#pragma unroll
        for (int t = 0; t < 8; ++t) {
            const __bf16* vr =
                VT + ((size_t)b * KVDIM + kvh * HEADDIM + t * 16 + ln) * NSEQ +
                s0 + half * 16;
            o[t] = wmma_bf16(pf, *(const v16bf*)vr, o[t]);
        }
    }

    // ---- normalize + store context (bf16) ----
#pragma unroll
    for (int j = 0; j < 8; ++j) lst[j] = 1.0f / lst[j];
#pragma unroll
    for (int t = 0; t < 8; ++t)
#pragma unroll
        for (int j = 0; j < 8; ++j) {
            int row = b * NSEQ + q0 + j + 8 * half;
            CTX[(size_t)row * DIMM + h * HEADDIM + t * 16 + ln] =
                (__bf16)(o[t][j] * lst[j]);
        }
}

// ---------------- host launcher ----------------
extern "C" void kernel_launch(void* const* d_in, const int* in_sizes, int n_in,
                              void* d_out, int out_size, void* d_ws,
                              size_t ws_size, hipStream_t stream) {
    (void)in_sizes; (void)n_in; (void)out_size; (void)ws_size;

    const float* q_in = (const float*)d_in[0];
    const float* k_in = (const float*)d_in[1];
    const float* v_in = (const float*)d_in[2];
    const float* Wq   = (const float*)d_in[3];
    const float* bq   = (const float*)d_in[4];
    const float* Wk   = (const float*)d_in[5];
    const float* bk   = (const float*)d_in[6];
    const float* Wv   = (const float*)d_in[7];
    const float* bv   = (const float*)d_in[8];
    const float* Wo   = (const float*)d_in[9];
    const float* bo   = (const float*)d_in[10];
    float* out = (float*)d_out;

    char* ws = (char*)d_ws;
    size_t off = 0;
    auto alloc = [&](size_t bytes) {
        char* p = ws + off;
        off += (bytes + 255) & ~(size_t)255;
        return p;
    };
    __bf16* qb  = (__bf16*)alloc((size_t)MROWS * DIMM * 2);   // bf16 inputs
    __bf16* kb  = (__bf16*)alloc((size_t)MROWS * KDIMM * 2);
    __bf16* vb  = (__bf16*)alloc((size_t)MROWS * KDIMM * 2);
    __bf16* Wqb = (__bf16*)alloc((size_t)DIMM * DIMM * 2);    // bf16 weights
    __bf16* Wkb = (__bf16*)alloc((size_t)KVDIM * KDIMM * 2);
    __bf16* Wvb = (__bf16*)alloc((size_t)KVDIM * KDIMM * 2);
    __bf16* Wob = (__bf16*)alloc((size_t)DIMM * DIMM * 2);
    __bf16* Qp  = (__bf16*)alloc((size_t)MROWS * DIMM * 2);   // roped Q
    __bf16* Kp  = (__bf16*)alloc((size_t)MROWS * KVDIM * 2);  // roped K
    __bf16* VTp = (__bf16*)alloc((size_t)MROWS * KVDIM * 2);  // V transposed
    __bf16* CTX = (__bf16*)alloc((size_t)MROWS * DIMM * 2);   // attn context

    // 1) fp32 -> bf16 conversions
    cvt_bf16<<<2048, 256, 0, stream>>>(q_in, qb, MROWS * DIMM);
    cvt_bf16<<<2048, 256, 0, stream>>>(k_in, kb, MROWS * KDIMM);
    cvt_bf16<<<2048, 256, 0, stream>>>(v_in, vb, MROWS * KDIMM);
    cvt_bf16<<<2048, 256, 0, stream>>>(Wq, Wqb, DIMM * DIMM);
    cvt_bf16<<<1024, 256, 0, stream>>>(Wk, Wkb, KVDIM * KDIMM);
    cvt_bf16<<<1024, 256, 0, stream>>>(Wv, Wvb, KVDIM * KDIMM);
    cvt_bf16<<<2048, 256, 0, stream>>>(Wo, Wob, DIMM * DIMM);

    // 2) projections (pipelined TDM-staged WMMA GEMMs, fused RoPE/transpose)
    gemm_kernel<MODE_Q><<<dim3(DIMM / 128, MROWS / 128), 128, 0, stream>>>(
        qb, Wqb, bq, Qp, DIMM, KDIMM);
    gemm_kernel<MODE_K><<<dim3(KVDIM / 128, MROWS / 128), 128, 0, stream>>>(
        kb, Wkb, bk, Kp, KVDIM, KDIMM);
    gemm_kernel<MODE_V><<<dim3(KVDIM / 128, MROWS / 128), 128, 0, stream>>>(
        vb, Wvb, bv, VTp, KVDIM, KDIMM);

    // 3) causal grouped-query flash attention
    attn_kernel<<<dim3(NSEQ / 64, QHEADS, BATCH), 128, 0, stream>>>(Qp, Kp,
                                                                    VTp, CTX);

    // 4) output projection -> f32 result
    gemm_kernel<MODE_O><<<dim3(DIMM / 128, MROWS / 128), 128, 0, stream>>>(
        CTX, Wob, bo, out, DIMM, DIMM);
}
